// Model_5171140625001
// MI455X (gfx1250) — matrix operand order repaired and checked by reading, not compiled, not run on hardware
//
#include <hip/hip_runtime.h>
#include <hip/hip_bf16.h>

#define B_ 128
#define T_ 256
#define H_ 1024
#define V_ 4096
#define RNN_BLOCKS 64   // 64 blocks x 8 waves = 512 waves = 8x64 16x16 tiles

typedef __bf16 bf16;
typedef __attribute__((ext_vector_type(16))) __bf16 v16bf;
typedef __attribute__((ext_vector_type(8)))  __bf16 v8bf;
typedef __attribute__((ext_vector_type(8)))  float  v8f;
typedef __attribute__((ext_vector_type(4)))  unsigned int u32x4;
typedef __attribute__((ext_vector_type(8)))  int i32x8;
typedef __attribute__((ext_vector_type(4)))  int i32x4;

// ---------------------------------------------------------------------------
// WMMA fragment loaders (bf16, 16x16x32, wave32) per CDNA5 ISA layouts.
// A (16xK row-major): lane<16 -> M=lane,    K = k0+{0..7, 16..23}
//                     lane>=16 -> M=lane-16, K = k0+{8..15, 24..31}
// ---------------------------------------------------------------------------
__device__ __forceinline__ v16bf load_a_frag(const bf16* __restrict__ A, int ld,
                                             int row0, int k0, int lane) {
    const int r    = row0 + (lane & 15);
    const int koff = k0 + ((lane >> 4) << 3);
    const bf16* p  = A + (size_t)r * ld + koff;
    v8bf lo = *(const v8bf*)(p);
    v8bf hi = *(const v8bf*)(p + 16);
    return __builtin_shufflevector(lo, hi, 0,1,2,3,4,5,6,7,8,9,10,11,12,13,14,15);
}

// B (KxN) supplied as W row-major [N][K] (we compute h @ W^T):
// per-lane column of W^T == K = k0+{0..7,16..23} (lane<16) / {8..15,24..31} of one W row.
__device__ __forceinline__ v16bf load_b_frag(const bf16* __restrict__ W, int ld,
                                             int n0, int k0, int lane) {
    const int n    = n0 + (lane & 15);
    const int koff = k0 + ((lane >> 4) << 3);
    const bf16* p  = W + (size_t)n * ld + koff;
    v8bf lo = *(const v8bf*)(p);
    v8bf hi = *(const v8bf*)(p + 16);
    return __builtin_shufflevector(lo, hi, 0,1,2,3,4,5,6,7,8,9,10,11,12,13,14,15);
}

__device__ __forceinline__ v8f wmma_bf16(v16bf a, v16bf b, v8f c) {
    return __builtin_amdgcn_wmma_f32_16x16x32_bf16(false, a, false, b,
                                                   (short)0, c, false, false);
}

// ---------------------------------------------------------------------------
// TDM: DMA a 2D bf16 tile (rows x cols, row stride = `stride0` elems) from
// global memory into LDS at byte offset `lds_off`. D# bitfields per ISA ch.8.
// Completion tracked by TENSORcnt. 6-arg builtin (clang-23 signature).
// ---------------------------------------------------------------------------
#if defined(__has_builtin)
#if __has_builtin(__builtin_amdgcn_tensor_load_to_lds) && \
    __has_builtin(__builtin_amdgcn_s_wait_tensorcnt)
#define USE_TDM 1
#endif
#endif

#ifdef USE_TDM
__device__ __forceinline__ void tdm_load_tile_to_lds(unsigned lds_off,
                                                     const void* gptr,
                                                     unsigned tile_d0,
                                                     unsigned tile_d1,
                                                     unsigned tensor_d0,
                                                     unsigned tensor_d1,
                                                     unsigned stride0) {
    const unsigned long long ga = (unsigned long long)(uintptr_t)gptr;
    u32x4 g0;
    g0[0] = 1u;                                   // count=1, user descriptor
    g0[1] = lds_off;                              // lds_addr (bytes)
    g0[2] = (unsigned)ga;                         // global_addr[31:0]
    g0[3] = (unsigned)(ga >> 32) | (2u << 30);    // global_addr[56:32] | type=2
    i32x8 g1;
    g1[0] = (int)(1u << 16);                      // wg_mask=0, data_size=1 (2B)
    g1[1] = (int)(tensor_d0 << 16);               // tensor_dim0[15:0] @bit48
    g1[2] = (int)((tensor_d0 >> 16) | (tensor_d1 << 16));
    g1[3] = (int)((tensor_d1 >> 16) | (tile_d0 << 16));   // tile_dim0 @bit112
    g1[4] = (int)(tile_d1 & 0xFFFFu);             // tile_dim1, tile_dim2=0
    g1[5] = (int)stride0;                         // tensor_dim0_stride[31:0]
    g1[6] = 0;                                    // stride0[47:32]=0, stride1 lo
    g1[7] = 0;                                    // stride1 hi (unused, 2D)
    i32x4 z4 = {0, 0, 0, 0};
    i32x8 z8 = {0, 0, 0, 0, 0, 0, 0, 0};
    __builtin_amdgcn_tensor_load_to_lds(g0, g1, z4, z4, z8, 0);
}
#endif

// ---------------------------------------------------------------------------
// Device-wide split barrier (persistent kernel). Monotonic counter in global
// scratch; target for barrier #ph is ph * gridDim.x.
// ---------------------------------------------------------------------------
__device__ __forceinline__ void grid_sync(unsigned* __restrict__ bar,
                                          unsigned nblk, unsigned ph) {
    __threadfence();      // release own stores to device scope
    __syncthreads();
    if (threadIdx.x == 0) {
        atomicAdd(bar, 1u);
        const unsigned tgt = ph * nblk;
        while (__atomic_load_n(bar, __ATOMIC_RELAXED) < tgt)
            __builtin_amdgcn_s_sleep(1);
    }
    __syncthreads();
    __threadfence();      // acquire: invalidate stale lines before reloads
}

// ---------------------------------------------------------------------------
// Prep kernels
// ---------------------------------------------------------------------------
__global__ void f32_to_bf16_kernel(const float* __restrict__ src,
                                   bf16* __restrict__ dst, int n) {
    for (int i = blockIdx.x * blockDim.x + threadIdx.x; i < n;
         i += gridDim.x * blockDim.x)
        dst[i] = (bf16)src[i];
}

__global__ void zero_bf16_kernel(bf16* __restrict__ p, int n) {
    int i = blockIdx.x * blockDim.x + threadIdx.x;
    if (i < n) p[i] = (bf16)0.0f;
}

// ---------------------------------------------------------------------------
// Persistent 2-layer RNN: all 256 timesteps in one launch.
// Each wave owns one 16x16 output tile (8 m-tiles x 64 n-tiles) for BOTH
// layers; device-wide barrier between layer-0 and layer-1 and between steps.
// ---------------------------------------------------------------------------
__global__ void rnn_persistent_kernel(const bf16* __restrict__ Whh0,
                                      const bf16* __restrict__ Wih1,
                                      const bf16* __restrict__ Whh1,
                                      const float* __restrict__ x,
                                      const float* __restrict__ w_ih0,
                                      const float* __restrict__ b_ih0,
                                      const float* __restrict__ b_hh0,
                                      const float* __restrict__ b_ih1,
                                      const float* __restrict__ b_hh1,
                                      bf16* __restrict__ h0buf,
                                      bf16* __restrict__ h1buf,
                                      bf16* __restrict__ h1all,
                                      float* __restrict__ h0_final,
                                      float* __restrict__ h1_final,
                                      unsigned* __restrict__ bar) {
    const int lane  = threadIdx.x & 31;
    const int wave  = threadIdx.x >> 5;
    const int gw    = blockIdx.x * 8 + wave;      // 0..511
    const int m0    = (gw >> 6) << 4;             // 8 m-tiles (B=128)
    const int n0    = (gw & 63) << 4;             // 64 n-tiles (H=1024)
    const int col   = n0 + (lane & 15);
    const int rbase = m0 + ((lane >> 4) << 3);
    const size_t BH = (size_t)B_ * H_;

    // Per-wave epilogue constants, hoisted out of the time loop.
    const float wv  = w_ih0[col];
    const float bb0 = b_ih0[col] + b_hh0[col];
    const float bb1 = b_ih1[col] + b_hh1[col];

    unsigned ph = 0;
    for (int t = 0; t < T_; ++t) {
        const int p = t & 1, q = p ^ 1;
        const bf16* h0p = h0buf + (size_t)p * BH;
        bf16*       h0q = h0buf + (size_t)q * BH;
        const bf16* h1p = h1buf + (size_t)p * BH;
        bf16*       h1q = h1buf + (size_t)q * BH;
        const int   last = (t == T_ - 1);

        // ---- layer 0: h0n = tanh(x_t*w_ih0 + b0 + h0 @ Whh0^T) ----
        v8f acc = {};
        #pragma unroll 4
        for (int k = 0; k < H_; k += 32) {
            v16bf a = load_a_frag(h0p,  H_, m0, k, lane);
            v16bf b = load_b_frag(Whh0, H_, n0, k, lane);
            acc = wmma_bf16(a, b, acc);
        }
        #pragma unroll
        for (int r = 0; r < 8; ++r) {
            const int   row = rbase + r;
            const float xv  = x[row * T_ + t];               // x is [B,1,T]
            const float v   = tanhf(acc[r] + xv * wv + bb0);
            h0q[(size_t)row * H_ + col] = (bf16)v;
            if (last) h0_final[(size_t)row * H_ + col] = v;
        }
        grid_sync(bar, gridDim.x, ++ph);

        // ---- layer 1: h1n = tanh(h0n @ Wih1^T + h1 @ Whh1^T + b1) ----
        acc = (v8f){};
        #pragma unroll 4
        for (int k = 0; k < H_; k += 32) {
            v16bf a = load_a_frag(h0q,  H_, m0, k, lane);
            v16bf b = load_b_frag(Wih1, H_, n0, k, lane);
            acc = wmma_bf16(a, b, acc);
        }
        #pragma unroll 4
        for (int k = 0; k < H_; k += 32) {
            v16bf a = load_a_frag(h1p,  H_, m0, k, lane);
            v16bf b = load_b_frag(Whh1, H_, n0, k, lane);
            acc = wmma_bf16(a, b, acc);
        }
        bf16* h1st = h1all + (size_t)t * BH;
        #pragma unroll
        for (int r = 0; r < 8; ++r) {
            const int   row = rbase + r;
            const float v   = tanhf(acc[r] + bb1);
            const bf16  vb  = (bf16)v;
            h1q[(size_t)row * H_ + col]  = vb;
            h1st[(size_t)row * H_ + col] = vb;
            if (last) h1_final[(size_t)row * H_ + col] = v;
        }
        grid_sync(bar, gridDim.x, ++ph);
    }
}

// ---------------------------------------------------------------------------
// Batched fc: out[b,t,v] = h1_all[t,b,:] @ W_fc^T + b_fc   (A = [T*B, H])
// Each wave computes a 32x64 block: 2 A-frags (LDS) x 4 B-frags -> 8 WMMA
// per k-step. A tile (32 x 1024 bf16, 64 KB) staged into LDS via the Tensor
// Data Mover (wave 0 issues one descriptor; TENSORcnt + barrier for arrival).
// grid (V/512, T*B/32) x 256 threads.
// ---------------------------------------------------------------------------
__global__ void fc_kernel(const bf16* __restrict__ A,
                          const bf16* __restrict__ Wfc,
                          const float* __restrict__ bfc,
                          float* __restrict__ out) {
    __shared__ __align__(16) bf16 ldsA[32 * H_];   // 64 KB, LDS offset 0

    const int tid    = threadIdx.x;
    const int lane   = tid & 31;
    const int wave   = tid >> 5;
    const int m0     = blockIdx.y << 5;                // 32 rows per block
    const int n_base = (blockIdx.x * 8 + wave) << 6;   // 64 cols per wave

    // Stage 32 consecutive A rows (64 KB, contiguous) into LDS.
#ifdef USE_TDM
    if (wave == 0) {
        tdm_load_tile_to_lds(/*lds_off=*/0u, A + (size_t)m0 * H_,
                             /*tile_d0=*/H_, /*tile_d1=*/32u,
                             /*tensor_d0=*/H_, /*tensor_d1=*/(unsigned)(T_ * B_),
                             /*stride0=*/H_);
        __builtin_amdgcn_s_wait_tensorcnt(0);
    }
    __syncthreads();
    // The TDM wrote ldsA behind the compiler's back: declare that memory
    // (in particular the LDS tile) has been modified so the ds_load path
    // is not undef-folded away.
    asm volatile("" : : "v"(&ldsA[0]) : "memory");
#else
    {
        const uint4* src = (const uint4*)(A + (size_t)m0 * H_);
        uint4*       dst = (uint4*)ldsA;
        #pragma unroll
        for (int i = tid; i < (32 * H_) / 8; i += 256) dst[i] = src[i];
    }
    __syncthreads();
#endif

    v8f a00 = {}, a01 = {}, a02 = {}, a03 = {};
    v8f a10 = {}, a11 = {}, a12 = {}, a13 = {};
    const bf16* wrow = Wfc + (size_t)(n_base + (lane & 15)) * H_;
    #pragma unroll 2
    for (int k = 0; k < H_; k += 32) {
        if (k + 64 < H_)
            __builtin_prefetch((const void*)(wrow + k + 64), 0, 0);
        v16bf af0 = load_a_frag(ldsA, H_, 0,  k, lane);   // LDS -> ds_load
        v16bf af1 = load_a_frag(ldsA, H_, 16, k, lane);
        v16bf b0  = load_b_frag(Wfc, H_, n_base,      k, lane);
        v16bf b1  = load_b_frag(Wfc, H_, n_base + 16, k, lane);
        v16bf b2  = load_b_frag(Wfc, H_, n_base + 32, k, lane);
        v16bf b3  = load_b_frag(Wfc, H_, n_base + 48, k, lane);
        a00 = wmma_bf16(af0, b0, a00);
        a01 = wmma_bf16(af0, b1, a01);
        a02 = wmma_bf16(af0, b2, a02);
        a03 = wmma_bf16(af0, b3, a03);
        a10 = wmma_bf16(af1, b0, a10);
        a11 = wmma_bf16(af1, b1, a11);
        a12 = wmma_bf16(af1, b2, a12);
        a13 = wmma_bf16(af1, b3, a13);
    }

    v8f accs[2][4] = {{a00, a01, a02, a03}, {a10, a11, a12, a13}};
    const int roff = (lane >> 4) << 3;
    #pragma unroll
    for (int half = 0; half < 2; ++half) {
        #pragma unroll
        for (int tile = 0; tile < 4; ++tile) {
            const int   col  = n_base + tile * 16 + (lane & 15);
            const float bias = bfc[col];
            #pragma unroll
            for (int r = 0; r < 8; ++r) {
                const int m  = m0 + half * 16 + roff + r;
                const int tt = m >> 7;        // m / B
                const int bb = m & 127;       // m % B
                out[((size_t)bb * T_ + tt) * V_ + col] = accs[half][tile][r] + bias;
            }
        }
    }
}

// ---------------------------------------------------------------------------
// Host launcher: 9 graph nodes total (4 converts, 3 zeros, RNN, FC).
// ---------------------------------------------------------------------------
extern "C" void kernel_launch(void* const* d_in, const int* in_sizes, int n_in,
                              void* d_out, int out_size, void* d_ws, size_t ws_size,
                              hipStream_t stream) {
    const float* x     = (const float*)d_in[0];
    const float* W_ih0 = (const float*)d_in[1];
    const float* W_hh0 = (const float*)d_in[2];
    const float* b_ih0 = (const float*)d_in[3];
    const float* b_hh0 = (const float*)d_in[4];
    const float* W_ih1 = (const float*)d_in[5];
    const float* W_hh1 = (const float*)d_in[6];
    const float* b_ih1 = (const float*)d_in[7];
    const float* b_hh1 = (const float*)d_in[8];
    const float* W_fc  = (const float*)d_in[9];
    const float* b_fc  = (const float*)d_in[10];
    float* out = (float*)d_out;

    const size_t HH = (size_t)H_ * H_;
    const size_t VH = (size_t)V_ * H_;
    const size_t BH = (size_t)B_ * H_;

    bf16* ws    = (bf16*)d_ws;
    bf16* Whh0b = ws;
    bf16* Wih1b = Whh0b + HH;
    bf16* Whh1b = Wih1b + HH;
    bf16* Wfcb  = Whh1b + HH;
    bf16* h0buf = Wfcb + VH;            // 2 * BH (ping-pong)
    bf16* h1buf = h0buf + 2 * BH;       // 2 * BH (ping-pong)
    bf16* h1all = h1buf + 2 * BH;       // T * B * H archive
    unsigned* bar = (unsigned*)(h1all + (size_t)T_ * BH);   // barrier counter

    f32_to_bf16_kernel<<<512, 256, 0, stream>>>(W_hh0, Whh0b, (int)HH);
    f32_to_bf16_kernel<<<512, 256, 0, stream>>>(W_ih1, Wih1b, (int)HH);
    f32_to_bf16_kernel<<<512, 256, 0, stream>>>(W_hh1, Whh1b, (int)HH);
    f32_to_bf16_kernel<<<2048, 256, 0, stream>>>(W_fc, Wfcb, (int)VH);
    zero_bf16_kernel<<<(int)((BH + 255) / 256), 256, 0, stream>>>(h0buf, (int)BH);
    zero_bf16_kernel<<<(int)((BH + 255) / 256), 256, 0, stream>>>(h1buf, (int)BH);
    zero_bf16_kernel<<<1, 256, 0, stream>>>((bf16*)bar, 2);   // counter = 0

    float* h0_final = out + (size_t)B_ * T_ * V_;
    float* h1_final = h0_final + BH;

    rnn_persistent_kernel<<<RNN_BLOCKS, 256, 0, stream>>>(
        Whh0b, Wih1b, Whh1b, x, W_ih0, b_ih0, b_hh0, b_ih1, b_hh1,
        h0buf, h1buf, h1all, h0_final, h1_final, bar);

    dim3 gFc(V_ / 512, (T_ * B_) / 32);   // (8, 1024)
    fc_kernel<<<gFc, 256, 0, stream>>>(h1all, Wfcb, b_fc, out);
}
